// HierarchicalPhimoeExperts_25580825215166
// MI455X (gfx1250) — matrix operand order repaired and checked by reading, not compiled, not run on hardware
//
#include <hip/hip_runtime.h>

#define T_DIM 2048
#define E_DIM 8
#define D_DIM 1024
#define F_DIM 4096
#define S_DIM 2
#define RB 16
#define RS 8

#define M_TILE 64
#define KCHUNK 128
#define WG_N   128   // 8 waves * 16 columns per workgroup

typedef __attribute__((ext_vector_type(16))) __bf16 bf16x16;
typedef __attribute__((ext_vector_type(8)))  float  v8f;

// ---------------------------------------------------------------- WMMA helpers

__device__ __forceinline__ v8f wmma_bf16(bf16x16 a, bf16x16 b, v8f c) {
  // 8 args: (neg_a, A, neg_b, B, c_mod, C, reuse_a, reuse_b)
  return __builtin_amdgcn_wmma_f32_16x16x32_bf16(false, a, false, b,
                                                 (short)0, c, false, false);
}

// A fragment (16x32 bf16): lane m (m=lane&15) holds row m; 16 contiguous
// K = koff..koff+15, koff = (lane<16) ? 0 : 16.
// p must point at row_base + koff (16B aligned).
__device__ __forceinline__ bf16x16 load_a_frag(const __bf16* p) {
  union { uint4 u[2]; bf16x16 v; } r;
  r.u[0] = *(const uint4*)(p);
  r.u[1] = *(const uint4*)(p + 8);
  return r.v;
}

// B fragment (32x16 bf16): lane n (n=lane&15) holds col n, 16 *contiguous*
// K values, K = kbase..kbase+15, kbase = (lane<16) ? 0 : 16. Source is fp32
// global memory; convert to bf16 (RNE) in registers.
__device__ __forceinline__ bf16x16 load_b_frag_f32(const float* p) {
  bf16x16 b;
  #pragma unroll
  for (int i = 0; i < 16; i += 4) {
    float4 f = *(const float4*)(p + i);
    b[i + 0] = (__bf16)f.x; b[i + 1] = (__bf16)f.y;
    b[i + 2] = (__bf16)f.z; b[i + 3] = (__bf16)f.w;
  }
  return b;
}

// ------------------------------------------- CDNA5 async global->LDS staging
// GLOBAL_LOAD_ASYNC_TO_LDS_B128: per-lane 16B DMA into LDS, tracked by
// ASYNCcnt (no VGPR data movement). The low 32 bits of a flat pointer into a
// __shared__ object are the wave-relative LDS byte offset (ISA aperture rule).

__device__ __forceinline__ void async_copy_b128(unsigned lds_off,
                                                const void* gaddr) {
  asm volatile("global_load_async_to_lds_b128 %0, %1, off"
               :: "v"(lds_off), "v"(gaddr)
               : "memory");
}

__device__ __forceinline__ void wait_async_all() {
  asm volatile("s_wait_asynccnt 0x0" ::: "memory");
}

// ---------------------------------------------------------------- tiny kernels

__global__ void moe_route(const long long* __restrict__ topi,
                          const float* __restrict__ tkw,
                          int* __restrict__ counts, int* __restrict__ tlist,
                          int* __restrict__ plist, float* __restrict__ wlist) {
  int p = blockIdx.x * 256 + threadIdx.x;           // pair id = t*2 + k
  if (p >= T_DIM * 2) return;
  int t = p >> 1;
  int e = (int)topi[p];
  int pos = atomicAdd(&counts[e], 1);
  tlist[e * T_DIM + pos] = t;
  plist[e * T_DIM + pos] = p;
  wlist[e * T_DIM + pos] = tkw[(size_t)t * E_DIM + e];
}

__global__ void moe_xcast(const float* __restrict__ x, __bf16* __restrict__ xb) {
  int i = (blockIdx.x * 256 + threadIdx.x) * 4;     // T*D divisible by 1024
  float4 f = *(const float4*)&x[i];
  union { __bf16 h[4]; uint2 u; } o;
  o.h[0] = (__bf16)f.x; o.h[1] = (__bf16)f.y;
  o.h[2] = (__bf16)f.z; o.h[3] = (__bf16)f.w;
  *(uint2*)&xb[i] = o.u;
}

// ------------------------------------------------ pass 1: gate/up + SiLU fuse

__global__ void __launch_bounds__(256) moe_gateup(
    const __bf16* __restrict__ xb,
    const float* __restrict__ Wg, const float* __restrict__ Wu,
    const int* __restrict__ counts, const int* __restrict__ tlist,
    const int* __restrict__ plist, const float* __restrict__ wlist,
    __bf16* __restrict__ hbuf) {
  const int e    = blockIdx.x >> 5;
  const int tile = blockIdx.x & 31;
  const int cnt  = counts[e];
  const int row0 = tile * M_TILE;
  if (row0 >= cnt) return;

  __shared__ __align__(16) __bf16 a_lds[2][M_TILE][KCHUNK];  // 32 KB, dbl-buf
  __shared__ int   s_pid[M_TILE];
  __shared__ int   s_src[M_TILE];
  __shared__ float s_w[M_TILE];

  const int tid  = threadIdx.x;
  const int wave = tid >> 5;
  const int lane = tid & 31;
  if (tid < M_TILE) {
    int r  = row0 + tid;
    bool v = r < cnt;
    s_src[tid] = v ? tlist[e * T_DIM + r] : 0;
    s_pid[tid] = v ? plist[e * T_DIM + r] : -1;
    s_w[tid]   = v ? wlist[e * T_DIM + r] : 0.0f;
  }

  // async DMA stage of the gathered 64-row x tile (bf16) into LDS buffer `buf`
  auto stage = [&](int buf, int k0) {
    for (int i = tid; i < M_TILE * (KCHUNK / 8); i += 256) {
      int m  = i >> 4;               // KCHUNK/8 == 16 chunks per row
      int kk = (i & 15) << 3;
      async_copy_b128((unsigned)(uintptr_t)&a_lds[buf][m][kk],
                      &xb[(size_t)s_src[m] * D_DIM + k0 + kk]);
    }
  };

  __syncthreads();                   // s_src visible before first stage
  stage(0, 0);
  wait_async_all();
  __syncthreads();

  const int nloc  = lane & 15;
  const int koff  = (lane < 16) ? 0 : 16;
  const int kbase = (lane < 16) ? 0 : 16;
  const int col   = blockIdx.y * WG_N + wave * 16 + nloc;   // F column
  const size_t wrow = ((size_t)e * F_DIM + col) * D_DIM;

  v8f accg[4] = {};
  v8f accu[4] = {};

  int cur = 0;
  for (int k0 = 0; k0 < D_DIM; k0 += KCHUNK) {
    if (k0 + KCHUNK < D_DIM) stage(cur ^ 1, k0 + KCHUNK);  // overlap DMA
    __builtin_prefetch(&Wg[wrow + k0 + KCHUNK], 0, 0);
    __builtin_prefetch(&Wu[wrow + k0 + KCHUNK], 0, 0);
    for (int kk = 0; kk < KCHUNK; kk += 32) {
      bf16x16 bg = load_b_frag_f32(&Wg[wrow + k0 + kk + kbase]);
      bf16x16 bu = load_b_frag_f32(&Wu[wrow + k0 + kk + kbase]);
      #pragma unroll
      for (int mt = 0; mt < 4; mt++) {
        bf16x16 a = load_a_frag(&a_lds[cur][mt * 16 + nloc][kk + koff]);
        accg[mt] = wmma_bf16(a, bg, accg[mt]);
        accu[mt] = wmma_bf16(a, bu, accu[mt]);
      }
    }
    wait_async_all();                // next-buffer DMA complete
    __syncthreads();                 // everyone done reading `cur`
    cur ^= 1;
  }

  // epilogue: h = silu(g) * u * routing_w, stored bf16 at [pair, F]
  const int mhi = (lane < 16) ? 0 : 8;
  #pragma unroll
  for (int mt = 0; mt < 4; mt++) {
    #pragma unroll
    for (int i = 0; i < 8; i++) {
      int m   = mt * 16 + mhi + i;
      int pid = s_pid[m];
      if (pid >= 0) {
        float g   = accg[mt][i], u = accu[mt][i];
        float sig = __builtin_amdgcn_rcpf(1.0f + __expf(-g));  // v_rcp_f32
        float h   = g * sig * u * s_w[m];
        hbuf[(size_t)pid * F_DIM + col] = (__bf16)h;
      }
    }
  }
}

// ------------------------------------------------------- pass 2: down project

__global__ void __launch_bounds__(256) moe_down(
    const __bf16* __restrict__ hbuf, const float* __restrict__ Wd,
    const int* __restrict__ counts, const int* __restrict__ plist,
    float* __restrict__ outp) {
  const int e    = blockIdx.x >> 5;
  const int tile = blockIdx.x & 31;
  const int cnt  = counts[e];
  const int row0 = tile * M_TILE;
  if (row0 >= cnt) return;

  __shared__ __align__(16) __bf16 a_lds[2][M_TILE][KCHUNK];  // 32 KB, dbl-buf
  __shared__ int s_pid[M_TILE];
  __shared__ int s_src[M_TILE];

  const int tid = threadIdx.x, wave = tid >> 5, lane = tid & 31;
  if (tid < M_TILE) {
    int r  = row0 + tid;
    bool v = r < cnt;
    int pid = v ? plist[e * T_DIM + r] : -1;
    s_pid[tid] = pid;
    s_src[tid] = (pid >= 0) ? pid : 0;
  }

  auto stage = [&](int buf, int k0) {
    for (int i = tid; i < M_TILE * (KCHUNK / 8); i += 256) {
      int m  = i >> 4;
      int kk = (i & 15) << 3;
      async_copy_b128((unsigned)(uintptr_t)&a_lds[buf][m][kk],
                      &hbuf[(size_t)s_src[m] * F_DIM + k0 + kk]);
    }
  };

  __syncthreads();
  stage(0, 0);
  wait_async_all();
  __syncthreads();

  const int nloc  = lane & 15;
  const int koff  = (lane < 16) ? 0 : 16;
  const int kbase = (lane < 16) ? 0 : 16;
  const int col   = blockIdx.y * WG_N + wave * 16 + nloc;   // D column
  const size_t wrow = ((size_t)e * D_DIM + col) * F_DIM;

  v8f acc[4] = {};

  int cur = 0;
  for (int k0 = 0; k0 < F_DIM; k0 += KCHUNK) {
    if (k0 + KCHUNK < F_DIM) stage(cur ^ 1, k0 + KCHUNK);
    __builtin_prefetch(&Wd[wrow + k0 + KCHUNK], 0, 0);
    for (int kk = 0; kk < KCHUNK; kk += 32) {
      bf16x16 b = load_b_frag_f32(&Wd[wrow + k0 + kk + kbase]);
      #pragma unroll
      for (int mt = 0; mt < 4; mt++) {
        bf16x16 a = load_a_frag(&a_lds[cur][mt * 16 + nloc][kk + koff]);
        acc[mt] = wmma_bf16(a, b, acc[mt]);
      }
    }
    wait_async_all();
    __syncthreads();
    cur ^= 1;
  }

  const int mhi = (lane < 16) ? 0 : 8;
  #pragma unroll
  for (int mt = 0; mt < 4; mt++) {
    #pragma unroll
    for (int i = 0; i < 8; i++) {
      int pid = s_pid[mt * 16 + mhi + i];
      if (pid >= 0) outp[(size_t)pid * D_DIM + col] = acc[mt][i];
    }
  }
}

// -------------------------------- pass 3: pair-sum + exact fp32 LoRA finalize

__global__ void __launch_bounds__(256) moe_finalize(
    const float* __restrict__ x, const long long* __restrict__ topi,
    const float* __restrict__ tkw,
    const float* __restrict__ A, const float* __restrict__ Bm,
    const float* __restrict__ sA, const float* __restrict__ sB,
    const float* __restrict__ sg,
    const float* __restrict__ outp, float* __restrict__ out) {
  const int t = blockIdx.x;
  const int tid = threadIdx.x, wave = tid >> 5, lane = tid & 31;
  const int NV = RB + S_DIM * (1 + RS);   // 34 reductions per expert

  __shared__ float xs[D_DIM];
  __shared__ float zv[2][NV];

  for (int d = tid; d < D_DIM; d += 256) xs[d] = x[(size_t)t * D_DIM + d];
  __syncthreads();

  // 68 wave-parallel dot products: base z (16), per-spawn gate (1) + zs (8)
  for (int v = wave; v < 2 * NV; v += 8) {
    int k  = v / NV, vi = v % NV;
    int e  = (int)topi[t * 2 + k];
    const float* wr;
    if (vi < RB) {
      wr = &A[((size_t)e * RB + vi) * D_DIM];
    } else {
      int s = (vi - RB) / (1 + RS), j = (vi - RB) % (1 + RS);
      wr = (j == 0) ? &sg[((size_t)e * S_DIM + s) * D_DIM]
                    : &sA[(((size_t)e * S_DIM + s) * RS + (j - 1)) * D_DIM];
    }
    float acc = 0.f;
    for (int d = lane; d < D_DIM; d += 32) acc += wr[d] * xs[d];
    #pragma unroll
    for (int off = 16; off; off >>= 1) acc += __shfl_xor(acc, off, 32);
    if (lane == 0) zv[k][vi] = acc;
  }
  __syncthreads();

  for (int d = tid; d < D_DIM; d += 256) {
    float r = outp[(size_t)(t * 2) * D_DIM + d] +
              outp[(size_t)(t * 2 + 1) * D_DIM + d];
    #pragma unroll
    for (int k = 0; k < 2; k++) {
      int e   = (int)topi[t * 2 + k];
      float w = tkw[(size_t)t * E_DIM + e];
      float c = 0.f;
      const float* br = &Bm[((size_t)e * D_DIM + d) * RB];
      #pragma unroll
      for (int rr = 0; rr < RB; rr++) c += br[rr] * zv[k][rr];
      r += 2.0f * w * c;                         // BASE_SCALE = 32/16 = 2
      #pragma unroll
      for (int s = 0; s < S_DIM; s++) {
        float gr = zv[k][RB + s * (1 + RS)];
        gr = gr > 0.f ? gr : 0.f;                // relu gate
        float cs = 0.f;
        const float* sbr = &sB[(((size_t)e * S_DIM + s) * D_DIM + d) * RS];
        #pragma unroll
        for (int rr = 0; rr < RS; rr++)
          cs += sbr[rr] * zv[k][RB + s * (1 + RS) + 1 + rr];
        r += 2.0f * w * gr * cs;                 // SPAWN_SCALE = 2
      }
    }
    out[(size_t)t * D_DIM + d] = r;
  }
}

// --------------------------------------------------------------------- launch

extern "C" void kernel_launch(void* const* d_in, const int* in_sizes, int n_in,
                              void* d_out, int out_size, void* d_ws, size_t ws_size,
                              hipStream_t stream) {
  (void)in_sizes; (void)n_in; (void)out_size; (void)ws_size;
  const float*     x    = (const float*)d_in[0];
  const long long* topi = (const long long*)d_in[1];
  const float*     tkw  = (const float*)d_in[2];
  const float*     Wg   = (const float*)d_in[3];
  const float*     Wu   = (const float*)d_in[4];
  const float*     Wd   = (const float*)d_in[5];
  const float*     A    = (const float*)d_in[6];
  const float*     Bm   = (const float*)d_in[7];
  const float*     sA   = (const float*)d_in[8];
  const float*     sB   = (const float*)d_in[9];
  const float*     sg   = (const float*)d_in[10];
  float* out = (float*)d_out;

  // workspace layout (all 4 KB aligned): counts | tlist | plist | wlist |
  //   xb (bf16 x, 4 MB) | hbuf (bf16 h[pair,F], 32 MB) | outp (f32, 16 MB)
  char* w = (char*)d_ws;
  int*    counts = (int*)(w);
  int*    tlist  = (int*)(w + 4096);
  int*    plist  = (int*)(w + 4096 + (1 << 16));
  float*  wlist  = (float*)(w + 4096 + 2 * (1 << 16));
  __bf16* xb     = (__bf16*)(w + 4096 + 3 * (1 << 16));
  __bf16* hbuf   = (__bf16*)(w + 4096 + 3 * (1 << 16)
                             + (size_t)T_DIM * D_DIM * 2);
  float*  outp   = (float*)(w + 4096 + 3 * (1 << 16)
                            + (size_t)T_DIM * D_DIM * 2
                            + (size_t)T_DIM * 2 * F_DIM * 2);

  hipMemsetAsync(counts, 0, E_DIM * sizeof(int), stream);
  moe_xcast<<<T_DIM * D_DIM / 1024, 256, 0, stream>>>(x, xb);
  moe_route<<<(T_DIM * 2 + 255) / 256, 256, 0, stream>>>(topi, tkw, counts,
                                                         tlist, plist, wlist);
  moe_gateup<<<dim3(E_DIM * 32, F_DIM / WG_N), 256, 0, stream>>>(
      xb, Wg, Wu, counts, tlist, plist, wlist, hbuf);
  moe_down<<<dim3(E_DIM * 32, D_DIM / WG_N), 256, 0, stream>>>(
      hbuf, Wd, counts, plist, outp);
  moe_finalize<<<T_DIM, 256, 0, stream>>>(x, topi, tkw, A, Bm, sA, sB, sg,
                                          outp, out);
}
